// SelfAttention_73400991089429
// MI455X (gfx1250) — matrix operand order repaired and checked by reading, not compiled, not run on hardware
//
#include <hip/hip_runtime.h>
#include <hip/hip_bf16.h>

typedef __attribute__((ext_vector_type(16))) _Float16 v16h;
typedef __attribute__((ext_vector_type(8)))  float    v8f;

#define TT 2048
#define CC 1024
#define HH 16
#define HD 64

// ---------------------------------------------------------------------------
// WMMA fragment helpers (CDNA5 wave32, V_WMMA_F32_16X16X32_F16)
// A (16x32 f16): lane m=lane&15, kb=16*(lane>>4); halves i = K kb+i
// B (32x16 f16): lane n=lane&15, kh=16*(lane>>4); halves i = K kh+i, col n
// C/D (16x16 f32): vgpr r -> row r+8*(lane>>4), col n=lane&15
// ---------------------------------------------------------------------------
__device__ __forceinline__ v16h load_a_frag(const _Float16* __restrict__ base,
                                            int lda, int lane) {
  const int m  = lane & 15;
  const int kb = (lane >> 4) << 4;
  const _Float16* p = base + m * lda + kb;
  v16h a;
#pragma unroll
  for (int i = 0; i < 16; ++i) a[i] = p[i];
  return a;
}

// Column n of B must be contiguous in memory: base + n*ldn + k  (so feed it a
// K-contiguous, N-strided ("transposed") operand).
__device__ __forceinline__ v16h load_b_frag(const _Float16* __restrict__ base,
                                            int ldn, int lane) {
  const int n  = lane & 15;
  const int kh = (lane >> 4) << 4;
  const _Float16* p = base + n * ldn + kh;
  v16h b;
#pragma unroll
  for (int i = 0; i < 16; ++i) b[i] = p[i];
  return b;
}

__device__ __forceinline__ v8f wmma16(v16h a, v16h b, v8f c) {
  return __builtin_amdgcn_wmma_f32_16x16x32_f16(false, a, false, b,
                                                (short)0, c, false, false);
}

__device__ __forceinline__ float rmax16(float v) {
  v = fmaxf(v, __shfl_xor(v, 1, 16));
  v = fmaxf(v, __shfl_xor(v, 2, 16));
  v = fmaxf(v, __shfl_xor(v, 4, 16));
  v = fmaxf(v, __shfl_xor(v, 8, 16));
  return v;
}
__device__ __forceinline__ float rsum16(float v) {
  v += __shfl_xor(v, 1, 16);
  v += __shfl_xor(v, 2, 16);
  v += __shfl_xor(v, 4, 16);
  v += __shfl_xor(v, 8, 16);
  return v;
}

// ---------------------------------------------------------------------------
// Prep kernels
// ---------------------------------------------------------------------------
__global__ void cvt_f32_f16(const float* __restrict__ in,
                            _Float16* __restrict__ out, int n) {
  int i = blockIdx.x * blockDim.x + threadIdx.x;
  if (i < n) out[i] = (_Float16)in[i];
}

// out[c][r] = (f16) in[r][c]   (rows x cols f32 -> cols x rows f16)
__global__ void transpose_f32_f16(const float* __restrict__ in,
                                  _Float16* __restrict__ out,
                                  int rows, int cols) {
  int i = blockIdx.x * blockDim.x + threadIdx.x;
  if (i < rows * cols) {
    int r = i / cols, c = i - r * cols;
    out[c * rows + r] = (_Float16)in[i];
  }
}

// ---------------------------------------------------------------------------
// GEMM mainloop: one wave computes a 64x64 tile (16 accumulators,
// 16 WMMAs per k=32 step, 8 fragment loads -> 2x the reuse of a 32x32 tile).
// ---------------------------------------------------------------------------
__device__ __forceinline__ void gemm_wave_64x64(
    const _Float16* __restrict__ A, const _Float16* __restrict__ BT,
    int row0, int col0, int K, int lane, v8f acc[4][4]) {
#pragma unroll
  for (int i = 0; i < 4; ++i)
#pragma unroll
    for (int j = 0; j < 4; ++j) acc[i][j] = {};

  for (int k = 0; k < K; k += 32) {
    v16h a[4], b[4];
#pragma unroll
    for (int i = 0; i < 4; ++i)
      a[i] = load_a_frag(A + (size_t)(row0 + 16 * i) * K + k, K, lane);
#pragma unroll
    for (int j = 0; j < 4; ++j)
      b[j] = load_b_frag(BT + (size_t)(col0 + 16 * j) * K + k, K, lane);
#pragma unroll
    for (int i = 0; i < 4; ++i)
#pragma unroll
      for (int j = 0; j < 4; ++j) acc[i][j] = wmma16(a[i], b[j], acc[i][j]);
  }
}

// ---------------------------------------------------------------------------
// GEMM: qkv = x(4096x1024) * w_qkv(1024x3072) + b_qkv  -> f16 qkv
// also scatters the V third transposed into vT[b,h,d,t] for the PV matmul.
// ---------------------------------------------------------------------------
__global__ __launch_bounds__(256) void gemm_qkv_kernel(
    const _Float16* __restrict__ A,   // M x K
    const _Float16* __restrict__ BT,  // N x K (transposed weights)
    const float* __restrict__ bias,   // N
    _Float16* __restrict__ Ch,        // M x N (f16)
    _Float16* __restrict__ vT,        // (B*H*HD) x T
    int M, int N, int K) {
  const int lane = threadIdx.x & 31;
  const int wid  = blockIdx.x * 8 + (threadIdx.x >> 5);
  const int tm_n = M >> 6, tn_n = N >> 6;
  if (wid >= tm_n * tn_n) return;
  const int row0 = (wid % tm_n) << 6;
  const int col0 = (wid / tm_n) << 6;

  v8f acc[4][4];
  gemm_wave_64x64(A, BT, row0, col0, K, lane, acc);

  const int n  = lane & 15;
  const int hi = lane >> 4;
#pragma unroll
  for (int i = 0; i < 4; ++i)
#pragma unroll
    for (int j = 0; j < 4; ++j) {
      const int rbase = row0 + 16 * i;
      const int col   = col0 + 16 * j + n;
#pragma unroll
      for (int r = 0; r < 8; ++r) {
        const int row = rbase + r + (hi << 3);
        float v = acc[i][j][r] + bias[col];
        _Float16 hv = (_Float16)v;
        Ch[(size_t)row * N + col] = hv;
        if (col >= 2 * CC) {  // V third -> transposed copy
          int cc = col - 2 * CC;
          int hh = cc >> 6, d = cc & 63;
          int bb = row >> 11, t = row & (TT - 1);
          vT[((size_t)((bb * HH + hh) << 6) + d) * TT + t] = hv;
        }
      }
    }
}

// ---------------------------------------------------------------------------
// GEMM: out = y(4096x1024) * w_out(1024x1024) + b_out -> f32
// ---------------------------------------------------------------------------
__global__ __launch_bounds__(256) void gemm_out_kernel(
    const _Float16* __restrict__ A, const _Float16* __restrict__ BT,
    const float* __restrict__ bias, float* __restrict__ Cf,
    int M, int N, int K) {
  const int lane = threadIdx.x & 31;
  const int wid  = blockIdx.x * 8 + (threadIdx.x >> 5);
  const int tm_n = M >> 6, tn_n = N >> 6;
  if (wid >= tm_n * tn_n) return;
  const int row0 = (wid % tm_n) << 6;
  const int col0 = (wid / tm_n) << 6;

  v8f acc[4][4];
  gemm_wave_64x64(A, BT, row0, col0, K, lane, acc);

  const int n  = lane & 15;
  const int hi = lane >> 4;
#pragma unroll
  for (int i = 0; i < 4; ++i)
#pragma unroll
    for (int j = 0; j < 4; ++j) {
      const int rbase = row0 + 16 * i;
      const int col   = col0 + 16 * j + n;
#pragma unroll
      for (int r = 0; r < 8; ++r) {
        const int row = rbase + r + (hi << 3);
        Cf[(size_t)row * N + col] = acc[i][j][r] + bias[col];
      }
    }
}

// ---------------------------------------------------------------------------
// Flash attention: one wave per 16-query tile of one (b,h); online softmax
// amortized over 64-key steps (4 S fragments, 8 S-WMMAs + 8 PV-WMMAs per
// step). P converts C-layout -> A-layout through a per-wave LDS buffer
// (DS ops are wave-ordered; wave_barrier stops compiler reordering).
// ---------------------------------------------------------------------------
__global__ __launch_bounds__(128) void flash_attn_kernel(
    const _Float16* __restrict__ qkvh,  // (B*T) x 3C
    const _Float16* __restrict__ vT,    // (B*H*HD) x T
    _Float16* __restrict__ yh) {        // (B*T) x C
  __shared__ _Float16 Pl[4][16 * 64];

  const int lane = threadIdx.x & 31;
  const int w    = threadIdx.x >> 5;
  const int bh   = blockIdx.y;
  const int b    = bh >> 4, h = bh & 15;
  const int qbase = (blockIdx.x * 4 + w) << 4;

  const int n  = lane & 15;
  const int hi = lane >> 4;

  const int ldq = 3 * CC;
  const _Float16* qptr = qkvh + (size_t)(b * TT + qbase) * ldq + h * HD;
  v16h qa0 = load_a_frag(qptr, ldq, lane);       // hd 0..31
  v16h qa1 = load_a_frag(qptr + 32, ldq, lane);  // hd 32..63

  const _Float16* kcol = qkvh + (size_t)(b * TT) * ldq + CC + h * HD;
  const _Float16* vcol = vT + (size_t)(bh * HD) * TT;

  v8f o[4] = {{}, {}, {}, {}};
  float mrow[8], lrow[8];
#pragma unroll
  for (int r = 0; r < 8; ++r) { mrow[r] = -1e30f; lrow[r] = 0.f; }

  const float scale = 0.125f;  // 1/sqrt(64)

  for (int j = 0; j <= qbase + 15; j += 64) {
    // S tile 16x64 = Q(16x64) @ K^T : four 16-key column tiles
    v8f s[4];
#pragma unroll
    for (int kt = 0; kt < 4; ++kt) {
      const _Float16* kb = kcol + (size_t)(j + 16 * kt) * ldq;
      v16h k0 = load_b_frag(kb, ldq, lane);
      v16h k1 = load_b_frag(kb + 32, ldq, lane);
      v8f z = {};
      s[kt] = wmma16(qa0, k0, z);
      s[kt] = wmma16(qa1, k1, s[kt]);
    }

    float pv[4][8];
#pragma unroll
    for (int r = 0; r < 8; ++r) {
      const int row = qbase + r + (hi << 3);
      float f[4], tm = -1e30f;
#pragma unroll
      for (int kt = 0; kt < 4; ++kt) {
        f[kt] = (j + 16 * kt + n <= row) ? s[kt][r] * scale : -1e30f;
        tm = fmaxf(tm, f[kt]);
      }
      tm = rmax16(tm);
      float mnew = fmaxf(mrow[r], tm);
      float a = __expf(mrow[r] - mnew);
      mrow[r] = mnew;
      float ts = 0.f;
#pragma unroll
      for (int kt = 0; kt < 4; ++kt) {
        float p = __expf(f[kt] - mnew);
        pv[kt][r] = p;
        ts += p;
      }
      lrow[r] = lrow[r] * a + rsum16(ts);
#pragma unroll
      for (int nt = 0; nt < 4; ++nt) o[nt][r] *= a;
    }

    // C-layout -> A-layout through LDS (same wave: DS ops stay ordered)
#pragma unroll
    for (int r = 0; r < 8; ++r) {
      int m = r + (hi << 3);
#pragma unroll
      for (int kt = 0; kt < 4; ++kt)
        Pl[w][m * 64 + 16 * kt + n] = (_Float16)pv[kt][r];
    }
    __builtin_amdgcn_wave_barrier();
    v16h pA0 = load_a_frag(&Pl[w][0], 64, lane);       // keys j..j+31
    v16h pA1 = load_a_frag(&Pl[w][0] + 32, 64, lane);  // keys j+32..j+63
    __builtin_amdgcn_wave_barrier();

    // O += P(16x64 keys) @ V(64keys x 64) : 4 hd column tiles x 2 k-steps
#pragma unroll
    for (int nt = 0; nt < 4; ++nt) {
      const _Float16* vb = vcol + (size_t)(16 * nt) * TT + j;
      v16h v0 = load_b_frag(vb, TT, lane);
      v16h v1 = load_b_frag(vb + 32, TT, lane);
      o[nt] = wmma16(pA0, v0, o[nt]);
      o[nt] = wmma16(pA1, v1, o[nt]);
    }
  }

#pragma unroll
  for (int r = 0; r < 8; ++r) {
    float inv = 1.0f / lrow[r];
    int row = qbase + r + (hi << 3);
    _Float16* yrow = yh + (size_t)(b * TT + row) * CC + h * HD;
#pragma unroll
    for (int nt = 0; nt < 4; ++nt)
      yrow[16 * nt + n] = (_Float16)(o[nt][r] * inv);
  }
}

// ---------------------------------------------------------------------------
// Host launcher
// ---------------------------------------------------------------------------
extern "C" void kernel_launch(void* const* d_in, const int* in_sizes, int n_in,
                              void* d_out, int out_size, void* d_ws,
                              size_t ws_size, hipStream_t stream) {
  (void)in_sizes; (void)n_in; (void)out_size; (void)ws_size;
  const float* x     = (const float*)d_in[0];
  const float* w_qkv = (const float*)d_in[1];
  const float* b_qkv = (const float*)d_in[2];
  const float* w_out = (const float*)d_in[3];
  const float* b_out = (const float*)d_in[4];
  float* out = (float*)d_out;

  // workspace layout (56 MiB total, f16 staging)
  char* ws = (char*)d_ws;
  _Float16* xh    = (_Float16*)(ws + 0);                        //  8 MiB
  _Float16* wqkvT = (_Float16*)(ws + (size_t)8  * 1024 * 1024); //  6 MiB
  _Float16* woutT = (_Float16*)(ws + (size_t)14 * 1024 * 1024); //  2 MiB
  _Float16* qkvh  = (_Float16*)(ws + (size_t)16 * 1024 * 1024); // 24 MiB
  _Float16* vT    = (_Float16*)(ws + (size_t)40 * 1024 * 1024); //  8 MiB
  _Float16* yh    = (_Float16*)(ws + (size_t)48 * 1024 * 1024); //  8 MiB

  const int M = 2 * TT;  // 4096

  cvt_f32_f16<<<(M * CC + 255) / 256, 256, 0, stream>>>(x, xh, M * CC);
  transpose_f32_f16<<<(CC * 3 * CC + 255) / 256, 256, 0, stream>>>(
      w_qkv, wqkvT, CC, 3 * CC);
  transpose_f32_f16<<<(CC * CC + 255) / 256, 256, 0, stream>>>(
      w_out, woutT, CC, CC);

  // qkv GEMM: 4096x3072x1024, 64x64 tiles -> (64*48)/8 = 384 blocks
  gemm_qkv_kernel<<<384, 256, 0, stream>>>(xh, wqkvT, b_qkv, qkvh, vT,
                                           M, 3 * CC, CC);

  // attention: grid (T/16/4, B*H)
  dim3 agrid(TT / 64, 2 * HH);
  flash_attn_kernel<<<agrid, 128, 0, stream>>>(qkvh, vT, yh);

  // out GEMM: 4096x1024x1024, 64x64 tiles -> (64*16)/8 = 128 blocks
  gemm_out_kernel<<<128, 256, 0, stream>>>(yh, woutT, b_out, out,
                                           M, CC, CC);
}
